// Attention_fusion_11682311045320
// MI455X (gfx1250) — matrix operand order repaired and checked by reading, not compiled, not run on hardware
//
#include <hip/hip_runtime.h>
#include <hip/hip_bf16.h>

typedef __attribute__((ext_vector_type(16))) _Float16 v16h;
typedef __attribute__((ext_vector_type(8)))  _Float16 v8h;
typedef __attribute__((ext_vector_type(8)))  float    v8f;
typedef __attribute__((ext_vector_type(4)))  float    f4;

#define BDIM 8
#define CDIM 128
#define HDIM 128
#define WDIM 128
#define NPIX (HDIM*WDIM)   // 16384
#define NHEADS 8
#define HD 16
#define NSPLIT 32

static __device__ __forceinline__ float sigmoidf_(float x) { return 1.0f / (1.0f + __expf(-x)); }

// ---------------------------------------------------------------------------
// Generic channel-major GEMM:
//   Out[m,n] = act( A[m,:]*diag(kscale) @ (X[:,n]*nscale[n]) + bias[m] + Cin[m,n] )
// A: MxK row-major (lda). X: KxN channel-major per batch (batch stride xbs).
// Block tile 32(M) x 256(N), K-step 32, 8 waves (wave32), f16 WMMA, f32 acc.
// Template flags keep the inner loop branch-free.
// A tile LDS: [32][40] f16 (row-major, k contiguous -> 2x ds_load_b128 / frag)
// B tile LDS: [256][40] f16 (column-major, k contiguous -> 2x ds_load_b128 / frag)
// ---------------------------------------------------------------------------
#define GTM 32
#define GTN 256
#define GTK 32

template<int ACT, bool KS, bool NS, bool HASB, bool HASC>
__global__ __launch_bounds__(256)
void gemm_cm_wmma(const float* __restrict__ A, int lda,
                  const float* __restrict__ X, long xbs,
                  const float* __restrict__ kscale, int ksb,
                  const float* __restrict__ nscale, long nsb,
                  const float* __restrict__ bias,
                  const float* __restrict__ Cin, long cbs,
                  float* __restrict__ Out, long obs,
                  int K, int N)
{
    __shared__ _Float16 As[GTM][GTK + 8];     // 32x40
    __shared__ _Float16 BsT[GTN][GTK + 8];    // 256x40 (column-major tile)

    const int tid  = threadIdx.x;
    const int b    = blockIdx.z;
    const int m0   = blockIdx.y * GTM;
    const int n0   = blockIdx.x * GTN;
    const int lane = tid & 31;
    const int wave = tid >> 5;
    const int wm   = wave >> 2;   // 0..1 -> 16-row slab
    const int wn   = wave & 3;    // 0..3 -> 64-col slab

    // per-thread column state for B staging (one column per thread)
    const float* xcol = X + (long)b * xbs + n0 + tid;
    float nsc = 1.0f;
    if (NS) nsc = nscale[(long)b * nsb + n0 + tid];

    v8f acc[4];
#pragma unroll
    for (int j = 0; j < 4; ++j) acc[j] = (v8f){0.f,0.f,0.f,0.f,0.f,0.f,0.f,0.f};

    for (int k0 = 0; k0 < K; k0 += GTK) {
        // ---- stage A tile (32x32, kscale folded in) ----
        {
            int m  = tid >> 3;
            int k4 = (tid & 7) * 4;
            f4 av = *(const f4*)(A + (long)(m0 + m) * lda + k0 + k4);
            if (KS) {
                const float* kp = kscale + (long)b * ksb + k0 + k4;
                av.x *= kp[0]; av.y *= kp[1]; av.z *= kp[2]; av.w *= kp[3];
            }
            As[m][k4 + 0] = (_Float16)av.x;
            As[m][k4 + 1] = (_Float16)av.y;
            As[m][k4 + 2] = (_Float16)av.z;
            As[m][k4 + 3] = (_Float16)av.w;
        }
        // ---- stage B tile: this thread's column, 32 k values ----
        {
            if (k0 + GTK < K) __builtin_prefetch(xcol + (long)(k0 + GTK) * N, 0, 1);
#pragma unroll
            for (int g = 0; g < 4; ++g) {
                v8h t;
#pragma unroll
                for (int i = 0; i < 8; ++i) {
                    float v = xcol[(long)(k0 + g * 8 + i) * N];
                    if (NS) v *= nsc;
                    t[i] = (_Float16)v;
                }
                *(v8h*)(&BsT[tid][g * 8]) = t;
            }
        }
        __syncthreads();

        // ---- A fragment: 16x32 f16 wave32 layout (two contiguous b128s) ----
        const int mrow = wm * 16 + (lane & 15);
        const int kb   = (lane < 16) ? 0 : 16;
        v8h a0 = *(const v8h*)(&As[mrow][kb]);
        v8h a1 = *(const v8h*)(&As[mrow][kb + 8]);
        v16h af = __builtin_shufflevector(a0, a1, 0,1,2,3,4,5,6,7,8,9,10,11,12,13,14,15);

        // ---- 4 B fragments (32x16 each) + WMMA ----
        const int kb2 = (lane < 16) ? 0 : 16;
#pragma unroll
        for (int j = 0; j < 4; ++j) {
            int col = wn * 64 + j * 16 + (lane & 15);
            v8h b0 = *(const v8h*)(&BsT[col][kb2]);
            v8h b1 = *(const v8h*)(&BsT[col][kb2 + 8]);
            v16h bf = __builtin_shufflevector(b0, b1, 0,1,2,3,4,5,6,7,8,9,10,11,12,13,14,15);
            acc[j] = __builtin_amdgcn_wmma_f32_16x16x32_f16(
                         false, af, false, bf, (short)0, acc[j], false, false);
        }
        __syncthreads();
    }

    // ---- epilogue ----
    const long ob = (long)b * obs;
    const long cb = (long)b * cbs;
#pragma unroll
    for (int j = 0; j < 4; ++j) {
        int col   = n0 + wn * 64 + j * 16 + (lane & 15);
        int rbase = m0 + wm * 16 + ((lane < 16) ? 0 : 8);
#pragma unroll
        for (int r = 0; r < 8; ++r) {
            int row = rbase + r;
            float v = acc[j][r];
            if (HASB) v += bias[row];
            if (HASC) v += Cin[cb + (long)row * N + col];
            if (ACT == 1) v = fmaxf(v, 0.0f);
            Out[ob + (long)row * N + col] = v;
        }
    }
}

// ---------------------------------------------------------------------------
// Global avg+max pool over HW for concat(x1,x2) channels.
// ---------------------------------------------------------------------------
__global__ __launch_bounds__(256)
void pool_avgmax(const float* __restrict__ x1, const float* __restrict__ x2,
                 float* __restrict__ avg, float* __restrict__ mx)
{
    __shared__ float ss[256], sm[256];
    int c = blockIdx.x;   // 0..255
    int b = blockIdx.y;
    const float* src = (c < CDIM) ? (x1 + ((long)b * CDIM + c) * NPIX)
                                  : (x2 + ((long)b * CDIM + (c - CDIM)) * NPIX);
    float s = 0.f, m = -3.4e38f;
    for (int i = threadIdx.x; i < NPIX; i += 256) { float v = src[i]; s += v; m = fmaxf(m, v); }
    ss[threadIdx.x] = s; sm[threadIdx.x] = m; __syncthreads();
    for (int st = 128; st > 0; st >>= 1) {
        if (threadIdx.x < st) {
            ss[threadIdx.x] += ss[threadIdx.x + st];
            sm[threadIdx.x] = fmaxf(sm[threadIdx.x], sm[threadIdx.x + st]);
        }
        __syncthreads();
    }
    if (threadIdx.x == 0) { avg[b * 256 + c] = ss[0] / (float)NPIX; mx[b * 256 + c] = sm[0]; }
}

// ---------------------------------------------------------------------------
// Channel-weight MLP: y(512) -> relu(W1 y + b1)(512) -> sigmoid(W2 h + b2)(256)
// ---------------------------------------------------------------------------
__global__ __launch_bounds__(256)
void cw_mlp(const float* __restrict__ avg, const float* __restrict__ mx,
            const float* __restrict__ w1, const float* __restrict__ b1,
            const float* __restrict__ w2, const float* __restrict__ b2,
            float* __restrict__ cw)
{
    __shared__ float ys[512], hs[512];
    int t = threadIdx.x;
    for (int b = 0; b < BDIM; ++b) {
        ys[t] = avg[b * 256 + t]; ys[256 + t] = mx[b * 256 + t];
        __syncthreads();
        for (int j = t; j < 512; j += 256) {
            float a = b1[j];
            const float* wr = w1 + (long)j * 512;
            for (int i = 0; i < 512; ++i) a += wr[i] * ys[i];
            hs[j] = fmaxf(a, 0.f);
        }
        __syncthreads();
        {
            float a = b2[t];
            const float* wr = w2 + (long)t * 512;
            for (int i = 0; i < 512; ++i) a += wr[i] * hs[i];
            cw[b * 256 + t] = sigmoidf_(a);
        }
        __syncthreads();
    }
}

// ---------------------------------------------------------------------------
// Spatial-weight final projection: s[b,n] = sigmoid(w2 . hidden[:,n] + b2)
// ---------------------------------------------------------------------------
__global__ __launch_bounds__(256)
void sw_scalar(const float* __restrict__ Hb, const float* __restrict__ w2,
               const float* __restrict__ b2, float* __restrict__ s)
{
    int b = blockIdx.y;
    long n = (long)blockIdx.x * 256 + threadIdx.x;
    const float* base = Hb + (long)b * CDIM * NPIX + n;
    float a = b2[0];
    for (int k = 0; k < CDIM; ++k) a += w2[k] * base[(long)k * NPIX];
    s[(long)b * NPIX + n] = sigmoidf_(a);
}

// ---------------------------------------------------------------------------
// Fused kv-projection + K.V^T partial reduction (per b,h,n-split).
// Never materializes the 134MB kv tensor.
// ---------------------------------------------------------------------------
__global__ __launch_bounds__(256)
void ctx_partial(const float* __restrict__ kvw, const float* __restrict__ cbuf,
                 float* __restrict__ ctxp)
{
    __shared__ float Kt[16][65];
    __shared__ float Vt[16][65];
    int split = blockIdx.x, h = blockIdx.y, b = blockIdx.z;
    int t = threadIdx.x;
    const int NC = NPIX / NSPLIT;   // 512
    int n0 = split * NC;
    const float* cb = cbuf + (long)b * CDIM * NPIX;
    int de_d = t >> 4, de_e = t & 15;
    float acc = 0.f;
    for (int nc = 0; nc < NC; nc += 64) {
#pragma unroll
        for (int i = 0; i < 8; ++i) {
            int e   = t + i * 256;
            int row = e >> 6;        // 0..31: rows 0..15 = K, 16..31 = V
            int col = e & 63;
            int wr  = (row < 16) ? (h * HD + row) : (CDIM + h * HD + (row - 16));
            const float* w  = kvw + (long)wr * CDIM;
            const float* xp = cb + n0 + nc + col;
            float a = 0.f;
            for (int k = 0; k < CDIM; ++k) a += w[k] * xp[(long)k * NPIX];
            if (row < 16) Kt[row][col] = a; else Vt[row - 16][col] = a;
        }
        __syncthreads();
#pragma unroll
        for (int col = 0; col < 64; ++col) acc += Kt[de_d][col] * Vt[de_e][col];
        __syncthreads();
    }
    ctxp[(((long)(b * NHEADS + h)) * NSPLIT + split) * 256 + t] = acc;
}

// ---------------------------------------------------------------------------
// Reduce partials, scale by hd^-0.5, softmax over d (axis -2).
// ---------------------------------------------------------------------------
__global__ __launch_bounds__(256)
void ctx_reduce_softmax(const float* __restrict__ ctxp, float* __restrict__ ctx)
{
    __shared__ float sm[256];
    __shared__ float cmax[16], csum[16];
    int bh = blockIdx.x;
    int t = threadIdx.x;
    float a = 0.f;
    for (int s = 0; s < NSPLIT; ++s) a += ctxp[((long)bh * NSPLIT + s) * 256 + t];
    a *= 0.25f;   // 16^-0.5
    sm[t] = a; __syncthreads();
    if (t < 16) {
        float m = -3.4e38f;
        for (int d = 0; d < 16; ++d) m = fmaxf(m, sm[d * 16 + t]);
        float s = 0.f;
        for (int d = 0; d < 16; ++d) s += __expf(sm[d * 16 + t] - m);
        cmax[t] = m; csum[t] = s;
    }
    __syncthreads();
    int e = t & 15;
    ctx[(long)bh * 256 + t] = __expf(sm[t] - cmax[e]) / csum[e];
}

// ---------------------------------------------------------------------------
// a[b, h*16+e, n] = sum_d q[b, h*16+d, n] * ctx[d,e]
// ---------------------------------------------------------------------------
__global__ __launch_bounds__(256)
void apply_ctx(const float* __restrict__ q, const float* __restrict__ ctx,
               float* __restrict__ out)
{
    __shared__ float cs[256];
    int h = blockIdx.y, b = blockIdx.z;
    long n = (long)blockIdx.x * 256 + threadIdx.x;
    cs[threadIdx.x] = ctx[((long)b * NHEADS + h) * 256 + threadIdx.x];
    __syncthreads();
    const float* qb = q + ((long)b * CDIM + h * HD) * NPIX + n;
    float qv[16];
#pragma unroll
    for (int d = 0; d < 16; ++d) qv[d] = qb[(long)d * NPIX];
    float* ob = out + ((long)b * CDIM + h * HD) * NPIX + n;
#pragma unroll
    for (int e = 0; e < 16; ++e) {
        float a = 0.f;
#pragma unroll
        for (int d = 0; d < 16; ++d) a += qv[d] * cs[d * 16 + e];
        ob[(long)e * NPIX] = a;
    }
}

// ---------------------------------------------------------------------------
// LayerNorm over channel dim, channel-major layout, in place.
// ---------------------------------------------------------------------------
__global__ __launch_bounds__(256)
void layernorm_cm(float* __restrict__ x, const float* __restrict__ g,
                  const float* __restrict__ bta)
{
    int b = blockIdx.y;
    long n = (long)blockIdx.x * 256 + threadIdx.x;
    float* base = x + (long)b * CDIM * NPIX + n;
    float s = 0.f, ss = 0.f;
    for (int c = 0; c < CDIM; ++c) { float v = base[(long)c * NPIX]; s += v; ss += v * v; }
    float m  = s / (float)CDIM;
    float var = ss / (float)CDIM - m * m;
    float rs = rsqrtf(var + 1e-5f);
    for (int c = 0; c < CDIM; ++c) {
        float v = base[(long)c * NPIX];
        base[(long)c * NPIX] = (v - m) * rs * g[c] + bta[c];
    }
}

// ---------------------------------------------------------------------------
// Depthwise 3x3 SAME conv + bias + relu (channel-major planes).
// ---------------------------------------------------------------------------
__global__ __launch_bounds__(256)
void dwconv3x3_relu(const float* __restrict__ in, const float* __restrict__ w,
                    const float* __restrict__ bias, float* __restrict__ out)
{
    int c = blockIdx.y, b = blockIdx.z;
    int pix = blockIdx.x * 256 + threadIdx.x;
    int y = pix >> 7, x = pix & 127;
    const float* ip = in + ((long)b * CDIM + c) * NPIX;
    const float* wc = w + c * 9;
    float a = bias[c];
#pragma unroll
    for (int ky = 0; ky < 3; ++ky) {
        int yy = y + ky - 1;
        if (yy < 0 || yy >= HDIM) continue;
#pragma unroll
        for (int kx = 0; kx < 3; ++kx) {
            int xx = x + kx - 1;
            if (xx < 0 || xx >= WDIM) continue;
            a += wc[ky * 3 + kx] * ip[yy * WDIM + xx];
        }
    }
    out[((long)b * CDIM + c) * NPIX + pix] = fmaxf(a, 0.f);
}

// ---------------------------------------------------------------------------
// Per-channel BN stats over (B, HW).
// ---------------------------------------------------------------------------
__global__ __launch_bounds__(256)
void bn_stats(const float* __restrict__ x, float* __restrict__ mean, float* __restrict__ var)
{
    __shared__ float ss[256], s2[256];
    int c = blockIdx.x;
    float s = 0.f, q = 0.f;
    for (int b = 0; b < BDIM; ++b) {
        const float* p = x + ((long)b * CDIM + c) * NPIX;
        for (int i = threadIdx.x; i < NPIX; i += 256) { float v = p[i]; s += v; q += v * v; }
    }
    ss[threadIdx.x] = s; s2[threadIdx.x] = q; __syncthreads();
    for (int st = 128; st > 0; st >>= 1) {
        if (threadIdx.x < st) { ss[threadIdx.x] += ss[threadIdx.x + st]; s2[threadIdx.x] += s2[threadIdx.x + st]; }
        __syncthreads();
    }
    if (threadIdx.x == 0) {
        float cnt = (float)BDIM * (float)NPIX;
        float m = ss[0] / cnt;
        mean[c] = m; var[c] = s2[0] / cnt - m * m;
    }
}

// R[i] += bn1(E[i])   (in place into R)
__global__ __launch_bounds__(256)
void add_bn(float* __restrict__ R, const float* __restrict__ E,
            const float* __restrict__ mean, const float* __restrict__ var,
            const float* __restrict__ g, const float* __restrict__ bt)
{
    long i = (long)blockIdx.x * 256 + threadIdx.x;
    int c = (int)((i / NPIX) % CDIM);
    float rs = rsqrtf(var[c] + 1e-5f);
    R[i] = R[i] + (E[i] - mean[c]) * rs * g[c] + bt[c];
}

// out[i] = bn2(Z[i])
__global__ __launch_bounds__(256)
void bn_apply(const float* __restrict__ Z,
              const float* __restrict__ mean, const float* __restrict__ var,
              const float* __restrict__ g, const float* __restrict__ bt,
              float* __restrict__ out)
{
    long i = (long)blockIdx.x * 256 + threadIdx.x;
    int c = (int)((i / NPIX) % CDIM);
    float rs = rsqrtf(var[c] + 1e-5f);
    out[i] = (Z[i] - mean[c]) * rs * g[c] + bt[c];
}

// ---------------------------------------------------------------------------
extern "C" void kernel_launch(void* const* d_in, const int* in_sizes, int n_in,
                              void* d_out, int out_size, void* d_ws, size_t ws_size,
                              hipStream_t stream)
{
    (void)in_sizes; (void)n_in; (void)out_size; (void)ws_size;
    const float* x1    = (const float*)d_in[0];
    const float* x2    = (const float*)d_in[1];
    const float* cw_w1 = (const float*)d_in[2];
    const float* cw_b1 = (const float*)d_in[3];
    const float* cw_w2 = (const float*)d_in[4];
    const float* cw_b2 = (const float*)d_in[5];
    const float* sw_w1 = (const float*)d_in[6];
    const float* sw_b1 = (const float*)d_in[7];
    const float* sw_w2 = (const float*)d_in[8];
    const float* sw_b2 = (const float*)d_in[9];
    const float* cp3_w = (const float*)d_in[10];
    const float* cp3_b = (const float*)d_in[11];
    const float* cp4_w = (const float*)d_in[12];
    const float* cp4_b = (const float*)d_in[13];
    const float* kv1_w = (const float*)d_in[14];
    const float* kv2_w = (const float*)d_in[15];
    const float* ep1_w = (const float*)d_in[16];
    const float* ep1_b = (const float*)d_in[17];
    const float* ep2_w = (const float*)d_in[18];
    const float* ep2_b = (const float*)d_in[19];
    const float* ln1_g = (const float*)d_in[20];
    const float* ln1_b = (const float*)d_in[21];
    const float* ln2_g = (const float*)d_in[22];
    const float* ln2_b = (const float*)d_in[23];
    const float* ce_res_w = (const float*)d_in[24];
    const float* ce1_w    = (const float*)d_in[25];
    const float* ce1_b    = (const float*)d_in[26];
    const float* ce_dw_w  = (const float*)d_in[27];
    const float* ce_dw_b  = (const float*)d_in[28];
    const float* ce2_w    = (const float*)d_in[29];
    const float* ce2_b    = (const float*)d_in[30];
    const float* bn1_g = (const float*)d_in[31];
    const float* bn1_b = (const float*)d_in[32];
    const float* bn2_g = (const float*)d_in[33];
    const float* bn2_b = (const float*)d_in[34];
    float* out = (float*)d_out;

    float* ws = (float*)d_ws;
    const size_t PLANE = (size_t)BDIM * CDIM * NPIX;   // 16.78M floats
    float* c1 = ws;                 // o1 reuses
    float* c2 = ws + PLANE;         // o2 reuses
    float* bA = ws + 2 * PLANE;     // sw-hidden / a1 / residual R / Z
    float* bB = ws + 3 * PLANE;     // a2 / E / E3
    float* smm   = ws + 4 * PLANE;
    float* avgp  = smm;               // 2048
    float* mxp   = avgp + 2048;       // 2048
    float* cwp   = mxp + 2048;        // 2048
    float* s1p   = cwp + 2048;        // 131072
    float* s2p   = s1p + 131072;      // 131072
    float* ctxp1 = s2p + 131072;      // 524288
    float* ctxp2 = ctxp1 + 524288;    // 524288
    float* ctx1  = ctxp2 + 524288;    // 16384
    float* ctx2  = ctx1 + 16384;      // 16384
    float* m1 = ctx2 + 16384; float* v1 = m1 + 128;
    float* m2 = v1 + 128;     float* v2 = m2 + 128;

    const long PB = (long)CDIM * NPIX;   // per-batch channel-plane stride
    dim3 blk(256);
    dim3 gg(NPIX / GTN, CDIM / GTM, BDIM);

    // 1) channel pooling + channel-weight MLP
    pool_avgmax<<<dim3(256, BDIM), blk, 0, stream>>>(x1, x2, avgp, mxp);
    cw_mlp<<<1, blk, 0, stream>>>(avgp, mxp, cw_w1, cw_b1, cw_w2, cw_b2, cwp);

    // 2) spatial weights: hidden = relu(sw_w1 @ x + b1) then sigmoid projection
    gemm_cm_wmma<1,false,false,true,false><<<gg, blk, 0, stream>>>(
        sw_w1, CDIM, x1, PB, nullptr, 0, nullptr, 0, sw_b1, nullptr, 0, bA, PB, CDIM, NPIX);
    sw_scalar<<<dim3(NPIX / 256, BDIM), blk, 0, stream>>>(bA, sw_w2, sw_b2, s1p);
    gemm_cm_wmma<1,false,false,true,false><<<gg, blk, 0, stream>>>(
        sw_w1, CDIM, x2, PB, nullptr, 0, nullptr, 0, sw_b1, nullptr, 0, bA, PB, CDIM, NPIX);
    sw_scalar<<<dim3(NPIX / 256, BDIM), blk, 0, stream>>>(bA, sw_w2, sw_b2, s2p);

    // 3) c1 = relu(cp3 @ (cw1*x1) + b), c2 = relu(cp4 @ (cw2*x2) + b)  (kscale on A)
    gemm_cm_wmma<1,true,false,true,false><<<gg, blk, 0, stream>>>(
        cp3_w, CDIM, x1, PB, cwp, 256, nullptr, 0, cp3_b, nullptr, 0, c1, PB, CDIM, NPIX);
    gemm_cm_wmma<1,true,false,true,false><<<gg, blk, 0, stream>>>(
        cp4_w, CDIM, x2, PB, cwp + CDIM, 256, nullptr, 0, cp4_b, nullptr, 0, c2, PB, CDIM, NPIX);

    // 4) fused kv projection + K.V^T reduction, then softmax over d
    ctx_partial<<<dim3(NSPLIT, NHEADS, BDIM), blk, 0, stream>>>(kv1_w, c1, ctxp1);
    ctx_partial<<<dim3(NSPLIT, NHEADS, BDIM), blk, 0, stream>>>(kv2_w, c2, ctxp2);
    ctx_reduce_softmax<<<dim3(BDIM * NHEADS), blk, 0, stream>>>(ctxp1, ctx1);
    ctx_reduce_softmax<<<dim3(BDIM * NHEADS), blk, 0, stream>>>(ctxp2, ctx2);

    // 5) a1 = q1 @ ctx2 (cross), a2 = q2 @ ctx1
    apply_ctx<<<dim3(NPIX / 256, NHEADS, BDIM), blk, 0, stream>>>(c1, ctx2, bA);
    apply_ctx<<<dim3(NPIX / 256, NHEADS, BDIM), blk, 0, stream>>>(c2, ctx1, bB);

    // 6) o = LN(x + ep[:, :C]@(s*x) + ep[:, C:]@a + b)   (split-K, nscale fuses s)
    gemm_cm_wmma<0,false,true,false,true><<<gg, blk, 0, stream>>>(
        ep1_w, 2 * CDIM, x1, PB, nullptr, 0, s1p, NPIX, nullptr, x1, PB, c1, PB, CDIM, NPIX);
    gemm_cm_wmma<0,false,false,true,true><<<gg, blk, 0, stream>>>(
        ep1_w + CDIM, 2 * CDIM, bA, PB, nullptr, 0, nullptr, 0, ep1_b, c1, PB, c1, PB, CDIM, NPIX);
    layernorm_cm<<<dim3(NPIX / 256, BDIM), blk, 0, stream>>>(c1, ln1_g, ln1_b);
    gemm_cm_wmma<0,false,true,false,true><<<gg, blk, 0, stream>>>(
        ep2_w, 2 * CDIM, x2, PB, nullptr, 0, s2p, NPIX, nullptr, x2, PB, c2, PB, CDIM, NPIX);
    gemm_cm_wmma<0,false,false,true,true><<<gg, blk, 0, stream>>>(
        ep2_w + CDIM, 2 * CDIM, bB, PB, nullptr, 0, nullptr, 0, ep2_b, c2, PB, c2, PB, CDIM, NPIX);
    layernorm_cm<<<dim3(NPIX / 256, BDIM), blk, 0, stream>>>(c2, ln2_g, ln2_b);

    // 7) residual = ce_res @ [o1;o2] -> bA ; E = ce1 @ [o1;o2] + b -> bB
    gemm_cm_wmma<0,false,false,false,false><<<gg, blk, 0, stream>>>(
        ce_res_w, 2 * CDIM, c1, PB, nullptr, 0, nullptr, 0, nullptr, nullptr, 0, bA, PB, CDIM, NPIX);
    gemm_cm_wmma<0,false,false,false,true><<<gg, blk, 0, stream>>>(
        ce_res_w + CDIM, 2 * CDIM, c2, PB, nullptr, 0, nullptr, 0, nullptr, bA, PB, bA, PB, CDIM, NPIX);
    gemm_cm_wmma<0,false,false,false,false><<<gg, blk, 0, stream>>>(
        ce1_w, 2 * CDIM, c1, PB, nullptr, 0, nullptr, 0, nullptr, nullptr, 0, bB, PB, CDIM, NPIX);
    gemm_cm_wmma<0,false,false,true,true><<<gg, blk, 0, stream>>>(
        ce1_w + CDIM, 2 * CDIM, c2, PB, nullptr, 0, nullptr, 0, ce1_b, bB, PB, bB, PB, CDIM, NPIX);

    // 8) depthwise 3x3 + relu (d_out as scratch), then ce2 projection
    dwconv3x3_relu<<<dim3(NPIX / 256, CDIM, BDIM), blk, 0, stream>>>(bB, ce_dw_w, ce_dw_b, out);
    gemm_cm_wmma<0,false,false,true,false><<<gg, blk, 0, stream>>>(
        ce2_w, CDIM, out, PB, nullptr, 0, nullptr, 0, ce2_b, nullptr, 0, bB, PB, CDIM, NPIX);

    // 9) BN chain: out = bn2(residual + bn1(e))
    bn_stats<<<dim3(CDIM), blk, 0, stream>>>(bB, m1, v1);
    add_bn<<<dim3((unsigned)(PLANE / 256)), blk, 0, stream>>>(bA, bB, m1, v1, bn1_g, bn1_b);
    bn_stats<<<dim3(CDIM), blk, 0, stream>>>(bA, m2, v2);
    bn_apply<<<dim3((unsigned)(PLANE / 256)), blk, 0, stream>>>(bA, m2, v2, bn2_g, bn2_b, out);
}
